// VectorQuantizer_8942121910392
// MI455X (gfx1250) — matrix operand order repaired and checked by reading, not compiled, not run on hardware
//
#include <hip/hip_runtime.h>
#include <hip/hip_bf16.h>
#include <math.h>

// ---------------------------------------------------------------------------
// VQ-VAE vector quantizer for MI455X (gfx1250, wave32, WMMA).
// Distances via split-precision bf16 WMMA (hi*hi + hi*lo + lo*hi ~= fp32).
// Codebook chunks double-buffered in LDS via GLOBAL_LOAD_ASYNC_TO_LDS_B128
// (ASYNCcnt), overlapping chunk k+1 staging with chunk k WMMA compute.
// ---------------------------------------------------------------------------

typedef __attribute__((ext_vector_type(16))) __bf16 v16bf;
typedef __attribute__((ext_vector_type(8)))  float  v8f;

union FragBF { uint4 u[2]; v16bf v; };

// ---- geometry ----
#define BDIM   32
#define CDIM   256
#define TDIM   2048
#define NROWS  65536          // 32*2048
#define NCODES 1024

#define ROWS_PER_BLOCK 128
#define CODE_CHUNK     64
#define NCHUNKS        (NCODES / CODE_CHUNK)   // 16
#define ZSTR           264    // 256 + 8 ushort pad (16B aligned rows, bank spread)
#define CB_HALF        (CODE_CHUNK * ZSTR)     // ushorts per hi (or lo) region
#define CB_BUF         (2 * CB_HALF)           // ushorts per double-buffer slot

// ---- d_out layout (floats) ----
#define ZQ_OFF    ((size_t)1)
#define PERP_OFF  ((size_t)16777217)
#define ME_OFF    ((size_t)16777218)
#define IDX_OFF   ((size_t)83886082)

// ---- workspace layout (bytes) ----
#define WS_CBH   ((size_t)0)         // 1024*256 ushort = 512KB
#define WS_CBL   ((size_t)524288)    // 512KB
#define WS_CNORM ((size_t)1048576)   // 1024 f32
#define WS_IDX   ((size_t)1052672)   // 65536 int
#define WS_HIST  ((size_t)1314816)   // 1024 int
#define WS_PART  ((size_t)1318912)   // 2048 f32

__device__ __forceinline__ unsigned short f2bf(float x) {
    unsigned u = __float_as_uint(x);
    unsigned r = u + 0x7FFFu + ((u >> 16) & 1u);   // round-to-nearest-even
    return (unsigned short)(r >> 16);
}
__device__ __forceinline__ float bf2f(unsigned short h) {
    return __uint_as_float(((unsigned)h) << 16);
}

__device__ __forceinline__ void wait_async_le16() {
#if __has_builtin(__builtin_amdgcn_s_wait_asynccnt)
    __builtin_amdgcn_s_wait_asynccnt(16);
#else
    asm volatile("s_wait_asynccnt 16" ::: "memory");
#endif
}
__device__ __forceinline__ void wait_async_0() {
#if __has_builtin(__builtin_amdgcn_s_wait_asynccnt)
    __builtin_amdgcn_s_wait_asynccnt(0);
#else
    asm volatile("s_wait_asynccnt 0" ::: "memory");
#endif
}

// issue one chunk's codebook hi/lo into an LDS buffer: 16 async b128 per wave
__device__ __forceinline__ void issue_cb_chunk(const unsigned short* __restrict__ cbh_g,
                                               const unsigned short* __restrict__ cbl_g,
                                               int chunk, int tid,
                                               unsigned ldsH, unsigned ldsL) {
    #pragma unroll
    for (int j = 0; j < 8; ++j) {
        const int idx = tid + j * 256;       // 2048 groups of 8 ushort (16B)
        const int n = idx >> 5;              // code row in chunk, 0..63
        const int g = idx & 31;              // 16B group in row
        const unsigned dH = ldsH + (unsigned)(n * ZSTR + g * 8) * 2u;
        const unsigned dL = ldsL + (unsigned)(n * ZSTR + g * 8) * 2u;
        const unsigned gOff = (unsigned)(((chunk * CODE_CHUNK + n) * CDIM + g * 8) * 2);
        asm volatile("global_load_async_to_lds_b128 %0, %1, %2 offset:0"
                     :: "v"(dH), "v"(gOff), "s"(cbh_g) : "memory");
        asm volatile("global_load_async_to_lds_b128 %0, %1, %2 offset:0"
                     :: "v"(dL), "v"(gOff), "s"(cbl_g) : "memory");
    }
}

// ---------------------------------------------------------------------------
// 1) codebook -> bf16 hi/lo split + ||e_k||^2 + zero histogram
// ---------------------------------------------------------------------------
__global__ void vq_prep(const float* __restrict__ cb,
                        unsigned short* __restrict__ cbh,
                        unsigned short* __restrict__ cbl,
                        float* __restrict__ cnorm,
                        int* __restrict__ hist) {
    __shared__ float red[256];
    const int k = blockIdx.x;
    const int c = threadIdx.x;
    float x = cb[(size_t)k * CDIM + c];
    unsigned short h = f2bf(x);
    float rem = x - bf2f(h);
    cbh[(size_t)k * CDIM + c] = h;
    cbl[(size_t)k * CDIM + c] = f2bf(rem);
    red[c] = x * x;
    __syncthreads();
    #pragma unroll
    for (int o = 128; o > 0; o >>= 1) {
        if (c < o) red[c] += red[c + o];
        __syncthreads();
    }
    if (c == 0) { cnorm[k] = red[0]; hist[k] = 0; }
}

// ---------------------------------------------------------------------------
// 2) zero the dense one-hot region (8B-aligned base -> float2 stores)
// ---------------------------------------------------------------------------
__global__ void vq_zero_me(float* __restrict__ out) {
    float2* p = (float2*)(out + ME_OFF);
    const size_t total = (size_t)NROWS * NCODES / 2;   // 33,554,432 float2
    const size_t stride = (size_t)gridDim.x * blockDim.x;
    for (size_t i = (size_t)blockIdx.x * blockDim.x + threadIdx.x; i < total; i += stride) {
        p[i] = make_float2(0.0f, 0.0f);
    }
}

// ---------------------------------------------------------------------------
// 3) WMMA distance + argmin. Block = 256 thr = 8 waves; 128-row strip.
//    LDS: z strip (hi/lo) + 2x codebook chunk buffers = 270,336 B.
// ---------------------------------------------------------------------------
extern __shared__ unsigned short smem[];

__global__ void vq_argmin(const float* __restrict__ z,
                          const unsigned short* __restrict__ cbh_g,
                          const unsigned short* __restrict__ cbl_g,
                          const float* __restrict__ cnorm,
                          int* __restrict__ idx_ws,
                          int* __restrict__ hist,
                          float* __restrict__ out) {
    unsigned short* zh = smem;
    unsigned short* zl = zh + ROWS_PER_BLOCK * ZSTR;
    unsigned short* cbuf = zl + ROWS_PER_BLOCK * ZSTR;   // 2 x (hi|lo) buffers

    const int tid  = threadIdx.x;
    const int lane = tid & 31;
    const int w    = tid >> 5;           // wave id 0..7
    const int r0   = blockIdx.x * ROWS_PER_BLOCK;
    const int b    = r0 >> 11;           // batch index (strip never crosses b)
    const int t0   = r0 & 2047;

    // LDS byte offsets of the two codebook buffers (low 32 bits of flat addr
    // of an LDS pointer == LDS offset)
    const unsigned cb0H = (unsigned)(size_t)(cbuf);
    const unsigned cb1H = (unsigned)(size_t)(cbuf + CB_BUF);

    // kick off chunk 0 staging before doing the z conversion work
    issue_cb_chunk(cbh_g, cbl_g, 0, tid, cb0H, cb0H + (unsigned)CB_HALF * 2u);

    // ---- stage z strip: global (b, c, t) -> LDS [t][c] as bf16 hi/lo ----
    const float* zg = z + ((size_t)b * CDIM) * TDIM + t0;
    for (int i = tid; i < ROWS_PER_BLOCK * CDIM; i += 256) {
        int c = i >> 7;                  // 0..255
        int t = i & 127;                 // 0..127 (lanes -> consecutive t: coalesced)
        float x = zg[(size_t)c * TDIM + t];
        unsigned short h = f2bf(x);
        zh[t * ZSTR + c] = h;
        zl[t * ZSTR + c] = f2bf(x - bf2f(h));
    }

    float minv[8];
    int   mini[8];
    #pragma unroll
    for (int v = 0; v < 8; ++v) { minv[v] = 3.4e38f; mini[v] = 0; }

    // fragment addressing: contiguous K order (k = 16*half + s), same for A and B
    const int lm     = lane & 15;
    const int hi16   = lane >> 4;
    const int rowLoc = w * 16 + lm;              // A: lane -> M row
    const int kbA    = hi16 ? 16 : 0;            // A: K 0-15 vs 16-31
    const int kbB    = hi16 ? 16 : 0;            // B: K 0-15 vs 16-31

    for (int chunk = 0; chunk < NCHUNKS; ++chunk) {
        const int cur = chunk & 1;
        // issue next chunk into the other buffer (previous iteration's
        // trailing barrier guarantees nobody is still reading it)
        if (chunk + 1 < NCHUNKS) {
            const unsigned nh = (cur ? cb0H : cb1H);
            issue_cb_chunk(cbh_g, cbl_g, chunk + 1, tid, nh, nh + (unsigned)CB_HALF * 2u);
            wait_async_le16();   // current chunk's 16 transfers retired (in-order)
        } else {
            wait_async_0();
        }
        __syncthreads();         // every wave's async data for `cur` is in LDS

        unsigned short* ch = cbuf + cur * CB_BUF;
        unsigned short* cl = ch + CB_HALF;

        // ---- GEMM tile: 16 rows x 64 codes, K=256, split-precision bf16 ----
        v8f acc[4];
        #pragma unroll
        for (int nt = 0; nt < 4; ++nt)
            acc[nt] = (v8f){0.f, 0.f, 0.f, 0.f, 0.f, 0.f, 0.f, 0.f};

        #pragma unroll
        for (int ks = 0; ks < 8; ++ks) {         // 8 x K32 steps
            const int c0 = ks * 32;
            FragBF a_h, a_l;
            const uint4* pah = (const uint4*)(zh + rowLoc * ZSTR + c0 + kbA);
            a_h.u[0] = pah[0]; a_h.u[1] = pah[1];        // 16 consecutive K
            const uint4* pal = (const uint4*)(zl + rowLoc * ZSTR + c0 + kbA);
            a_l.u[0] = pal[0]; a_l.u[1] = pal[1];
            #pragma unroll
            for (int nt = 0; nt < 4; ++nt) {
                const int nLoc = nt * 16 + lm;           // B: lane -> N column
                FragBF b_h, b_l;
                const uint4* pbh = (const uint4*)(ch + nLoc * ZSTR + c0 + kbB);
                b_h.u[0] = pbh[0]; b_h.u[1] = pbh[1];    // 16 consecutive K
                const uint4* pbl = (const uint4*)(cl + nLoc * ZSTR + c0 + kbB);
                b_l.u[0] = pbl[0]; b_l.u[1] = pbl[1];
                v8f cc = acc[nt];
                cc = __builtin_amdgcn_wmma_f32_16x16x32_bf16(false, a_h.v, false, b_h.v,
                                                             (short)0, cc, false, false);
                cc = __builtin_amdgcn_wmma_f32_16x16x32_bf16(false, a_h.v, false, b_l.v,
                                                             (short)0, cc, false, false);
                cc = __builtin_amdgcn_wmma_f32_16x16x32_bf16(false, a_l.v, false, b_h.v,
                                                             (short)0, cc, false, false);
                acc[nt] = cc;
            }
        }

        // ---- d_k = ||e_k||^2 - 2 z.e_k  (||z||^2 cancels in argmin) ----
        #pragma unroll
        for (int nt = 0; nt < 4; ++nt) {
            const int ncode = chunk * CODE_CHUNK + nt * 16 + lm;
            const float cn = cnorm[ncode];
            #pragma unroll
            for (int v = 0; v < 8; ++v) {
                float d = cn - 2.0f * acc[nt][v];
                if (d < minv[v]) { minv[v] = d; mini[v] = ncode; }
            }
        }
        __syncthreads();   // done reading buf[cur]; next iter may refill it
    }

    // ---- cross-lane argmin over the 16 codes per C-matrix half ----
    #pragma unroll
    for (int v = 0; v < 8; ++v) {
        float mv = minv[v];
        int   mi = mini[v];
        #pragma unroll
        for (int off = 8; off > 0; off >>= 1) {
            float ov = __shfl_xor(mv, off, 32);
            int   oi = __shfl_xor(mi, off, 32);
            if (ov < mv || (ov == mv && oi < mi)) { mv = ov; mi = oi; }
        }
        minv[v] = mv; mini[v] = mi;
    }

    // lanes 0 / 16 own rows M=0..7 / 8..15 of this wave's tile
    if (lm == 0) {
        #pragma unroll
        for (int v = 0; v < 8; ++v) {
            const int row = r0 + w * 16 + hi16 * 8 + v;
            const int k = mini[v];
            idx_ws[row] = k;
            out[IDX_OFF + (size_t)row] = (float)k;
            atomicAdd(&hist[k], 1);
            out[ME_OFF + (size_t)row * NCODES + (size_t)k] = 1.0f;
        }
    }
}

// ---------------------------------------------------------------------------
// 4) z_q gather (straight-through => z_q exactly) + loss partial sums
// ---------------------------------------------------------------------------
__global__ void vq_gather_loss(const float* __restrict__ z,
                               const float* __restrict__ cb,
                               const int* __restrict__ idx_ws,
                               float* __restrict__ out,
                               float* __restrict__ partials) {
    __shared__ float red[256];
    float s = 0.0f;
    const size_t total = (size_t)BDIM * CDIM * TDIM;   // 16,777,216 in z layout
    const size_t stride = (size_t)gridDim.x * blockDim.x;
    for (size_t i = (size_t)blockIdx.x * blockDim.x + threadIdx.x; i < total; i += stride) {
        const int t  = (int)(i & 2047);
        const int bc = (int)(i >> 11);                 // b*256 + c
        const int c  = bc & 255;
        const int bb = bc >> 8;
        const int n  = (bb << 11) | t;
        const int k  = idx_ws[n];
        const float q = cb[(size_t)k * CDIM + c];
        const float d = q - z[i];
        s += d * d;
        out[ZQ_OFF + i] = q;                           // z_q_out, (B,C,T) layout
    }
    red[threadIdx.x] = s;
    __syncthreads();
    #pragma unroll
    for (int o = 128; o > 0; o >>= 1) {
        if (threadIdx.x < o) red[threadIdx.x] += red[threadIdx.x + o];
        __syncthreads();
    }
    if (threadIdx.x == 0) partials[blockIdx.x] = red[0];
}

// ---------------------------------------------------------------------------
// 5) finalize: loss = 1.25 * mean(sq); perplexity from histogram
// ---------------------------------------------------------------------------
__global__ void vq_finalize(const int* __restrict__ hist,
                            const float* __restrict__ partials,
                            float* __restrict__ out) {
    __shared__ float red[1024];
    const int tid = threadIdx.x;
    // deterministic fixed-order sum of 2048 partials
    red[tid] = partials[tid] + partials[tid + 1024];
    __syncthreads();
    #pragma unroll
    for (int o = 512; o > 0; o >>= 1) {
        if (tid < o) red[tid] += red[tid + o];
        __syncthreads();
    }
    if (tid == 0) out[0] = 1.25f * red[0] / 16777216.0f;
    __syncthreads();
    // perplexity
    const float e = (float)hist[tid] * (1.0f / 65536.0f);
    red[tid] = e * logf(e + 1e-10f);
    __syncthreads();
    #pragma unroll
    for (int o = 512; o > 0; o >>= 1) {
        if (tid < o) red[tid] += red[tid + o];
        __syncthreads();
    }
    if (tid == 0) out[PERP_OFF] = expf(-red[0]);
}

// ---------------------------------------------------------------------------
extern "C" void kernel_launch(void* const* d_in, const int* in_sizes, int n_in,
                              void* d_out, int out_size, void* d_ws, size_t ws_size,
                              hipStream_t stream) {
    (void)in_sizes; (void)n_in; (void)out_size; (void)ws_size;
    const float* z  = (const float*)d_in[0];   // (32, 256, 2048) fp32
    const float* cb = (const float*)d_in[1];   // (1024, 256) fp32
    float* out = (float*)d_out;
    char*  ws  = (char*)d_ws;

    unsigned short* cbh   = (unsigned short*)(ws + WS_CBH);
    unsigned short* cbl   = (unsigned short*)(ws + WS_CBL);
    float*          cnorm = (float*)(ws + WS_CNORM);
    int*            idxb  = (int*)(ws + WS_IDX);
    int*            hist  = (int*)(ws + WS_HIST);
    float*          part  = (float*)(ws + WS_PART);

    vq_prep<<<NCODES, 256, 0, stream>>>(cb, cbh, cbl, cnorm, hist);
    vq_zero_me<<<4096, 256, 0, stream>>>(out);

    const size_t smem_bytes =
        (size_t)(2 * ROWS_PER_BLOCK * ZSTR + 2 * CB_BUF) * sizeof(unsigned short);
    vq_argmin<<<NROWS / ROWS_PER_BLOCK, 256, smem_bytes, stream>>>(
        z, cbh, cbl, cnorm, idxb, hist, out);

    vq_gather_loss<<<2048, 256, 0, stream>>>(z, cb, idxb, out, part);
    vq_finalize<<<1, 1024, 0, stream>>>(hist, part, out);
}
